// UNet_ConvLSTM_76716705841759
// MI455X (gfx1250) — matrix operand order repaired and checked by reading, not compiled, not run on hardware
//
#include <hip/hip_runtime.h>

// ---------------------------------------------------------------------------
// CDNA5 (gfx1250) UNet-ConvLSTM. Dominant compute = 3x3 convs lowered to
// implicit GEMM on v_wmma_f32_16x16x32_bf16 (wave32).
// Block = 8 waves sharing one 64-channel weight block, staged through LDS
// (double-buffered, 4KB/stage) via GLOBAL_LOAD_ASYNC_TO_LDS_B128 (ASYNCcnt)
// when available, else global_load_b128 + ds_store_b128.
// Each wave: 64(M) x 32(N) tile = 4 M-subtiles x 2 N-subtiles
//  -> 8 WMMA per 32-wide K step; A fragments: 2x ds_load_b128 each.
// B side: branchless clamped f32 gather + cvt to bf16.
// Requires Ktot%32==0, Cout%64==0, (P/32)%8==0 (true for every layer except
// the 3-channel stem conv, which runs on a direct kernel).
// ---------------------------------------------------------------------------

typedef __attribute__((ext_vector_type(16))) __bf16 bf16x16;
typedef __attribute__((ext_vector_type(8)))  float  f32x8;
typedef __attribute__((ext_vector_type(4)))  int    i32x4;

#if __has_builtin(__builtin_amdgcn_global_load_async_to_lds_b128) && \
    __has_builtin(__builtin_amdgcn_s_wait_asynccnt)
#define USE_ASYNC_LDS 1
#else
#define USE_ASYNC_LDS 0
#endif

__device__ __forceinline__ float sigm(float x) { return 1.0f / (1.0f + __expf(-x)); }

// ----------------- weight prep: f32 (Cout,Cin,3,3) -> bf16 [Cout][K], K=Cin*9,
// BN scale g*rsqrt(1+1e-5) folded per output channel -----------------
__global__ void wprep_k(const float* __restrict__ W, const float* __restrict__ g,
                        __bf16* __restrict__ O, int Cout, int K) {
  long i = (long)blockIdx.x * blockDim.x + threadIdx.x;
  long total = (long)Cout * K;
  if (i >= total) return;
  int co = (int)(i / K);
  float s = g ? g[co] * 0.9999950000375f : 1.0f;
  O[i] = (__bf16)(W[i] * s);
}

// ----------------- implicit-GEMM 3x3 pad1 conv via WMMA bf16 -----------------
__global__ __launch_bounds__(256) void conv3x3_wmma(
    const float* __restrict__ X, const __bf16* __restrict__ Wb,
    const float* __restrict__ bias, float* __restrict__ O,
    int Bn, int Cin, int H, int Wd, int Cout, int relu)
{
  const int Ktot = Cin * 9;            // multiple of 32
  const int HW = H * Wd;
  const int P = Bn * HW;               // multiple of 256
  const int Mblk = Cout >> 6;          // 64-output-channel blocks
  const int lane = threadIdx.x & 31;
  const int wib  = threadIdx.x >> 5;   // wave-in-block 0..7

  // block owns one weight block (tm); its 8 waves take 8 consecutive N tiles
  const int tm  = blockIdx.x % Mblk;
  const int tn  = (blockIdx.x / Mblk) * 8 + wib;
  const int co0 = tm << 6;
  const int n0  = tn << 5;
  const int m = lane & 15, hi = lane >> 4;

  // LDS staging: 64 rows x 32 K of bf16 = 4KB per stage, double buffered
  __shared__ uint4 smem[2][256];       // [stage][row*4 + part]
  const int frow  = threadIdx.x >> 2;  // 0..63
  const int fpart = threadIdx.x & 3;   // 0..3 (8 bf16 each)
  const __bf16* fsrc = Wb + (size_t)(co0 + frow) * Ktot;
  auto fill = [&](int buf, int k0) {
    const uint4* src = (const uint4*)(fsrc + k0) + fpart;
#if USE_ASYNC_LDS
    // direct memory->LDS DMA, tracked by ASYNCcnt
    __builtin_amdgcn_global_load_async_to_lds_b128(
        (__attribute__((address_space(1))) i32x4*)(void*)src,
        (__attribute__((address_space(3))) i32x4*)(void*)&smem[buf][threadIdx.x],
        0, 0);
#else
    __builtin_prefetch((const char*)src + 64, 0, 3);   // next k-step
    smem[buf][threadIdx.x] = *src;
#endif
  };
  auto stage_sync = [&]() {
#if USE_ASYNC_LDS
    __builtin_amdgcn_s_wait_asynccnt(0);   // this wave's async LDS writes done
#endif
    __syncthreads();                        // cross-wave visibility
  };

  // two pixels per lane (C/D layout: N = lane&15; N-subtiles 16 apart)
  int p0 = n0 + m, p1 = p0 + 16;
  int b0 = p0 / HW, q0 = p0 % HW, y0 = q0 / Wd, x0 = q0 % Wd;
  int b1 = p1 / HW, q1 = p1 % HW, y1 = q1 / Wd, x1 = q1 % Wd;
  const float* Xb0 = X + (size_t)b0 * Cin * HW;
  const float* Xb1 = X + (size_t)b1 * Cin * HW;

  // branchless B-fragment gather: contiguous k order, same as A (lane half: K=hi*16+j)
  auto gatherB = [&](const float* __restrict__ Xb, int yc, int xc, int k0) -> bf16x16 {
    bf16x16 bb;
    int kb = k0 + (hi << 4);
    int ci = kb / 9, r = kb % 9;
    #pragma unroll
    for (int j = 0; j < 16; ++j) {
      int dy = r / 3;
      int dx = r - dy * 3;
      int yy = yc + dy - 1, xx = xc + dx - 1;
      unsigned ok = ((unsigned)yy < (unsigned)H) & ((unsigned)xx < (unsigned)Wd);
      int yyc = ok ? yy : 0, xxc = ok ? xx : 0;
      float v = Xb[((size_t)ci * H + yyc) * Wd + xxc];   // unconditional load
      bb[j] = (__bf16)(ok ? v : 0.0f);                   // select, no branch
      ++r; int c = (r == 9); ci += c; r = c ? 0 : r;
    }
    return bb;
  };

  union AF { uint4 q[2]; bf16x16 v; };
  f32x8 acc0[4] = {}, acc1[4] = {};
  fill(0, 0);
  stage_sync();
  int cb = 0;
  for (int k0 = 0; k0 < Ktot; k0 += 32) {
    if (k0 + 32 < Ktot) fill(1 - cb, k0 + 32);   // prefetch next stage
    bf16x16 bf0 = gatherB(Xb0, y0, x0, k0);
    bf16x16 bf1 = gatherB(Xb1, y1, x1, k0);
    // hoist all A fragments (8x ds_load_b128, one wait), then 8 WMMA
    AF a[4];
    #pragma unroll
    for (int s = 0; s < 4; ++s) {
      const uint4* srow = &smem[cb][(((s << 4) + m) << 2)];
      a[s].q[0] = srow[2 * hi];      // K = k0 + hi*16 .. +7
      a[s].q[1] = srow[2 * hi + 1];  // K = k0 + hi*16 + 8 .. +7
    }
    #pragma unroll
    for (int s = 0; s < 4; ++s) {
      acc0[s] = __builtin_amdgcn_wmma_f32_16x16x32_bf16(false, a[s].v, false, bf0,
                                                        (short)0, acc0[s], false, false);
      acc1[s] = __builtin_amdgcn_wmma_f32_16x16x32_bf16(false, a[s].v, false, bf1,
                                                        (short)0, acc1[s], false, false);
    }
    stage_sync();                  // stage 1-cb filled; all reads of cb done
    cb ^= 1;
  }

  size_t ob0 = (size_t)b0 * Cout * HW + (size_t)y0 * Wd + x0;
  size_t ob1 = (size_t)b1 * Cout * HW + (size_t)y1 * Wd + x1;
  #pragma unroll
  for (int s = 0; s < 4; ++s) {
    #pragma unroll
    for (int r = 0; r < 8; ++r) {
      int co = co0 + (s << 4) + r + (hi << 3);   // C layout: M = r + 8*hi
      float bz = bias[co];
      float v0 = acc0[s][r] + bz;
      float v1 = acc1[s][r] + bz;
      if (relu) { v0 = fmaxf(v0, 0.0f); v1 = fmaxf(v1, 0.0f); }
      O[ob0 + (size_t)co * HW] = v0;
      O[ob1 + (size_t)co * HW] = v1;
    }
  }
}

// ----------------- direct conv for the 3-channel stem (K=27) -----------------
__global__ void conv3x3_small(const float* __restrict__ X, const float* __restrict__ Wt,
                              const float* __restrict__ g, const float* __restrict__ bias,
                              float* __restrict__ O, int Bn, int Cin, int H, int Wd, int Cout) {
  long i = (long)blockIdx.x * blockDim.x + threadIdx.x;
  long total = (long)Bn * Cout * H * Wd;
  if (i >= total) return;
  int x = (int)(i % Wd); long t = i / Wd;
  int y = (int)(t % H);  t /= H;
  int co = (int)(t % Cout);
  int b  = (int)(t / Cout);
  const float* Xb = X + (size_t)b * Cin * H * Wd;
  float s = 0.0f;
  for (int ci = 0; ci < Cin; ++ci)
    #pragma unroll
    for (int dy = 0; dy < 3; ++dy)
      #pragma unroll
      for (int dx = 0; dx < 3; ++dx) {
        int yy = y + dy - 1, xx = x + dx - 1;
        unsigned ok = ((unsigned)yy < (unsigned)H) & ((unsigned)xx < (unsigned)Wd);
        int yyc = ok ? yy : 0, xxc = ok ? xx : 0;
        float v = Xb[((size_t)ci * H + yyc) * Wd + xxc];
        s += Wt[((co * Cin + ci) * 3 + dy) * 3 + dx] * (ok ? v : 0.0f);
      }
  float sc = g[co] * 0.9999950000375f;
  O[i] = fmaxf(s * sc + bias[co], 0.0f);
}

// ----------------- 1x1 conv (tiny: Cout=5) -----------------
__global__ void conv1x1_k(const float* __restrict__ X, const float* __restrict__ W,
                          const float* __restrict__ bias, float* __restrict__ O,
                          int Bn, int Cin, int H, int Wd, int Cout) {
  long i = (long)blockIdx.x * blockDim.x + threadIdx.x;
  long total = (long)Bn * Cout * H * Wd;
  if (i >= total) return;
  int x = (int)(i % Wd); long t = i / Wd;
  int y = (int)(t % H);  t /= H;
  int co = (int)(t % Cout);
  int b  = (int)(t / Cout);
  const float* Xb = X + ((size_t)b * Cin * H + y) * Wd + x;
  float s = bias[co];
  for (int ci = 0; ci < Cin; ++ci) s += W[co * Cin + ci] * Xb[(size_t)ci * H * Wd];
  O[i] = s;
}

// ----------------- 2x2 max pool -----------------
__global__ void pool2_k(const float* __restrict__ X, float* __restrict__ O,
                        int Bn, int C, int H, int Wd) {
  int Ho = H >> 1, Wo = Wd >> 1;
  long i = (long)blockIdx.x * blockDim.x + threadIdx.x;
  long total = (long)Bn * C * Ho * Wo;
  if (i >= total) return;
  int ox = (int)(i % Wo); long t = i / Wo;
  int oy = (int)(t % Ho); t /= Ho;
  int c  = (int)(t % C);
  int b  = (int)(t / C);
  const float* Xb = X + ((size_t)b * C + c) * H * Wd + (size_t)(2 * oy) * Wd + 2 * ox;
  O[i] = fmaxf(fmaxf(Xb[0], Xb[1]), fmaxf(Xb[Wd], Xb[Wd + 1]));
}

// ----------------- bilinear 2x upsample, pos = i*(n-1)/(2n-1) -----------------
__global__ void up2x_k(const float* __restrict__ X, float* __restrict__ O,
                       int Bn, int C, int H, int Wd) {
  int Ho = 2 * H, Wo = 2 * Wd;
  long i = (long)blockIdx.x * blockDim.x + threadIdx.x;
  long total = (long)Bn * C * Ho * Wo;
  if (i >= total) return;
  int ox = (int)(i % Wo); long t = i / Wo;
  int oy = (int)(t % Ho); t /= Ho;
  int c  = (int)(t % C);
  int b  = (int)(t / C);
  float py = oy * ((float)(H - 1) / (float)(2 * H - 1));
  int y0 = (int)py; float ty = py - (float)y0;
  int y1 = y0 + 1; if (y1 > H - 1) y1 = H - 1;
  float px = ox * ((float)(Wd - 1) / (float)(2 * Wd - 1));
  int x0 = (int)px; float tx = px - (float)x0;
  int x1 = x0 + 1; if (x1 > Wd - 1) x1 = Wd - 1;
  const float* Xb = X + ((size_t)b * C + c) * H * Wd;
  float v00 = Xb[(size_t)y0 * Wd + x0], v01 = Xb[(size_t)y0 * Wd + x1];
  float v10 = Xb[(size_t)y1 * Wd + x0], v11 = Xb[(size_t)y1 * Wd + x1];
  float a0 = v00 * (1.0f - ty) + v10 * ty;
  float a1 = v01 * (1.0f - ty) + v11 * ty;
  O[i] = a0 * (1.0f - tx) + a1 * tx;
}

// ----------------- generic copies -----------------
__global__ void copyk(const float* __restrict__ s, float* __restrict__ d, long n) {
  long i = (long)blockIdx.x * blockDim.x + threadIdx.x;
  if (i < n) d[i] = s[i];
}
__global__ void fillz(float* __restrict__ d, long n) {
  long i = (long)blockIdx.x * blockDim.x + threadIdx.x;
  if (i < n) d[i] = 0.0f;
}
__global__ void copy_chan(const float* __restrict__ src, float* __restrict__ dst,
                          int Bn, int Csrc, int Cdst, int coff, long HW) {
  long i = (long)blockIdx.x * blockDim.x + threadIdx.x;
  long total = (long)Bn * Csrc * HW;
  if (i >= total) return;
  long pos = i % HW; long t = i / HW;
  int c = (int)(t % Csrc);
  int b = (int)(t / Csrc);
  dst[((size_t)b * Cdst + coff + c) * HW + pos] = src[i];
}
__global__ void permute_k(const float* __restrict__ X, float* __restrict__ F,
                          int B, int T, long CHW) {
  long i = (long)blockIdx.x * blockDim.x + threadIdx.x;
  long total = (long)B * T * CHW;
  if (i >= total) return;
  long chw = i % CHW; long t2 = i / CHW;
  int t = (int)(t2 % T);
  int b = (int)(t2 / T);
  F[((size_t)t * B + b) * CHW + chw] = X[i];
}

// ----------------- message-passing scan step (in-place safe) -----------------
__global__ void mp_step(const float* __restrict__ Xin, float* __restrict__ Y,
                        const float* __restrict__ Kw,
                        int Bn, int C, int H, int Wd, int vertical, int cur, int prev) {
  int L = vertical ? Wd : H;
  long i = (long)blockIdx.x * blockDim.x + threadIdx.x;
  long total = (long)Bn * C * L;
  if (i >= total) return;
  int pos = (int)(i % L); long t = i / L;
  int co = (int)(t % C);
  int b  = (int)(t / C);
  auto IDX = [&](int c, int strip, int q) -> size_t {
    return vertical ? (((size_t)b * C + c) * H + strip) * Wd + q
                    : (((size_t)b * C + c) * H + q) * Wd + strip;
  };
  float v = Xin[IDX(co, cur, pos)];
  if (prev >= 0) {
    float s = 0.0f;
    for (int ci = 0; ci < C; ++ci) {
      const float* kr = Kw + ((size_t)co * C + ci) * 9;
      #pragma unroll
      for (int tt = 0; tt < 9; ++tt) {
        int q = pos + tt - 4;
        unsigned ok = (unsigned)q < (unsigned)L;
        float pv = Y[IDX(ci, prev, ok ? q : 0)];
        s += kr[tt] * (ok ? pv : 0.0f);
      }
    }
    v += fmaxf(s, 0.0f);
  }
  Y[IDX(co, cur, pos)] = v;
}

// ----------------- ConvLSTM pointwise gate update -----------------
__global__ void lstm_point(const float* __restrict__ G, float* __restrict__ Hs,
                           float* __restrict__ Cs, float* __restrict__ HsOut,
                           int Bn, int hd, long HW) {
  long i = (long)blockIdx.x * blockDim.x + threadIdx.x;
  long total = (long)Bn * hd * HW;
  if (i >= total) return;
  long pos = i % HW; long t = i / HW;
  int ch = (int)(t % hd);
  int b  = (int)(t / hd);
  size_t gb = (size_t)b * 4 * hd * HW;
  float gi = G[gb + (size_t)ch * HW + pos];
  float gf = G[gb + (size_t)(hd + ch) * HW + pos];
  float go = G[gb + (size_t)(2 * hd + ch) * HW + pos];
  float gg = G[gb + (size_t)(3 * hd + ch) * HW + pos];
  float c = Cs[i];
  c = sigm(gf) * c + sigm(gi) * tanhf(gg);
  float h = sigm(go) * tanhf(c);
  Cs[i] = c; Hs[i] = h; HsOut[i] = h;
}

// ---------------------------------------------------------------------------
extern "C" void kernel_launch(void* const* d_in, const int* in_sizes, int n_in,
                              void* d_out, int out_size, void* d_ws, size_t ws_size,
                              hipStream_t stream) {
  (void)in_sizes; (void)n_in; (void)out_size; (void)ws_size;
  const float* X = (const float*)d_in[0];
  auto W = [&](int i) { return (const float*)d_in[i]; };
  auto NB = [](long n) -> unsigned { return (unsigned)((n + 255) / 256); };

  // ---- workspace bump allocator (256B aligned) ----
  char* base = (char*)d_ws;
  size_t off = 0;
  auto alloc_f = [&](size_t n) -> float* {
    float* p = (float*)(base + off); off = (off + n * 4 + 255) & ~(size_t)255; return p; };
  auto alloc_h = [&](size_t n) -> __bf16* {
    __bf16* p = (__bf16*)(base + off); off = (off + n * 2 + 255) & ~(size_t)255; return p; };

  // activations (f32); batch-10 = (t,b) major so x5 == LSTM seq, last frame = slice [8:10]
  float* fFrames = alloc_f(983040);        // (10,3,128,256)
  float* fT64    = alloc_f(20971520);      // dconv intermediate full-res
  float* fX1     = alloc_f(20971520);      // (10,64,128,256) pre-mp skip
  float* fM      = alloc_f(20971520);      // mp scans (in-place)
  float* fPool   = alloc_f(5242880);
  float* fTmp    = alloc_f(10485760);
  float* fX2     = alloc_f(10485760);      // (10,128,64,128)
  float* fX3     = alloc_f(5242880);       // (10,256,32,64)
  float* fX4     = alloc_f(2621440);       // (10,512,16,32)
  float* fX5     = alloc_f(655360);        // (10,512,8,16) == (5,2,512,8,16)
  float* fH      = alloc_f(131072);
  float* fC      = alloc_f(131072);
  float* fCat    = alloc_f(262144);        // (2,1024,8,16)
  float* fGates  = alloc_f(524288);        // (2,2048,8,16)
  float* fHs0    = alloc_f(655360);
  float* fHs1    = alloc_f(655360);
  float* fUp     = alloc_f(4194304);
  float* fU      = alloc_f(4194304);
  float* fCatD   = alloc_f(8388608);

  // bf16 weights [Cout][Cin*9], BN scale folded
  __bf16* wInc2 = alloc_h(64 * 576);
  __bf16* wD1a  = alloc_h(128 * 576);      __bf16* wD1b  = alloc_h(128 * 1152);
  __bf16* wD2a  = alloc_h(256 * 1152);     __bf16* wD2b  = alloc_h(256 * 2304);
  __bf16* wD3a  = alloc_h(512 * 2304);     __bf16* wD3b  = alloc_h(512 * 4608);
  __bf16* wD4a  = alloc_h(512 * 4608);     __bf16* wD4b  = alloc_h(512 * 4608);
  __bf16* wU1a  = alloc_h(256 * 9216);     __bf16* wU1b  = alloc_h(256 * 2304);
  __bf16* wU2a  = alloc_h(128 * 4608);     __bf16* wU2b  = alloc_h(128 * 1152);
  __bf16* wU3a  = alloc_h(64 * 2304);      __bf16* wU3b  = alloc_h(64 * 576);
  __bf16* wU4a  = alloc_h(64 * 1152);      __bf16* wU4b  = alloc_h(64 * 576);
  __bf16* wL0   = alloc_h((size_t)2048 * 9216);
  __bf16* wL1   = alloc_h((size_t)2048 * 9216);

  auto prep = [&](int wi, int gi, __bf16* dst, int co, int k) {
    long n = (long)co * k;
    wprep_k<<<NB(n), 256, 0, stream>>>(W(wi), gi >= 0 ? W(gi) : nullptr, dst, co, k);
  };
  prep(4, 5, wInc2, 64, 576);
  prep(7, 8, wD1a, 128, 576);   prep(10, 11, wD1b, 128, 1152);
  prep(13, 14, wD2a, 256, 1152);prep(16, 17, wD2b, 256, 2304);
  prep(19, 20, wD3a, 512, 2304);prep(22, 23, wD3b, 512, 4608);
  prep(25, 26, wD4a, 512, 4608);prep(28, 29, wD4b, 512, 4608);
  prep(31, 32, wU1a, 256, 9216);prep(34, 35, wU1b, 256, 2304);
  prep(37, 38, wU2a, 128, 4608);prep(40, 41, wU2b, 128, 1152);
  prep(43, 44, wU3a, 64, 2304); prep(46, 47, wU3b, 64, 576);
  prep(49, 50, wU4a, 64, 1152); prep(52, 53, wU4b, 64, 576);
  prep(59, -1, wL0, 2048, 9216);prep(61, -1, wL1, 2048, 9216);

  auto conv3 = [&](const float* Xi, const __bf16* Wb, const float* bias, float* O,
                   int Bn, int Cin, int H, int Wd, int Cout, int relu) {
    // blocks = Mblk * Nblk/8 ; each block = 8 waves sharing one weight block
    unsigned blocks = (unsigned)((long)(Cout >> 6) * (((long)Bn * H * Wd) >> 8));
    conv3x3_wmma<<<blocks, 256, 0, stream>>>(Xi, Wb, bias, O, Bn, Cin, H, Wd, Cout, relu);
  };

  // ---- encoder (batch 10 = 5 frames x 2) ----
  permute_k<<<NB(983040), 256, 0, stream>>>(X, fFrames, 2, 5, 3L * 32768);
  conv3x3_small<<<NB(20971520), 256, 0, stream>>>(fFrames, W(1), W(2), W(3), fT64,
                                                  10, 3, 128, 256, 64);
  conv3(fT64, wInc2, W(6), fX1, 10, 64, 128, 256, 64, 1);

  // message passing: copy x1, then 4 sequential in-place scans
  copyk<<<NB(20971520), 256, 0, stream>>>(fX1, fM, 20971520);
  for (int cur = 1; cur < 128; ++cur)
    mp_step<<<640, 256, 0, stream>>>(fM, fM, W(55), 10, 64, 128, 256, 1, cur, cur - 1);
  for (int cur = 126; cur >= 0; --cur)
    mp_step<<<640, 256, 0, stream>>>(fM, fM, W(56), 10, 64, 128, 256, 1, cur, cur + 1);
  for (int cur = 1; cur < 256; ++cur)
    mp_step<<<320, 256, 0, stream>>>(fM, fM, W(57), 10, 64, 128, 256, 0, cur, cur - 1);
  for (int cur = 254; cur >= 0; --cur)
    mp_step<<<320, 256, 0, stream>>>(fM, fM, W(58), 10, 64, 128, 256, 0, cur, cur + 1);

  pool2_k<<<NB(5242880), 256, 0, stream>>>(fM, fPool, 10, 64, 128, 256);
  conv3(fPool, wD1a, W(9), fTmp, 10, 64, 64, 128, 128, 1);
  conv3(fTmp, wD1b, W(12), fX2, 10, 128, 64, 128, 128, 1);

  pool2_k<<<NB(2621440), 256, 0, stream>>>(fX2, fPool, 10, 128, 64, 128);
  conv3(fPool, wD2a, W(15), fTmp, 10, 128, 32, 64, 256, 1);
  conv3(fTmp, wD2b, W(18), fX3, 10, 256, 32, 64, 256, 1);

  pool2_k<<<NB(1310720), 256, 0, stream>>>(fX3, fPool, 10, 256, 32, 64);
  conv3(fPool, wD3a, W(21), fTmp, 10, 256, 16, 32, 512, 1);
  conv3(fTmp, wD3b, W(24), fX4, 10, 512, 16, 32, 512, 1);

  pool2_k<<<NB(655360), 256, 0, stream>>>(fX4, fPool, 10, 512, 16, 32);
  conv3(fPool, wD4a, W(27), fTmp, 10, 512, 8, 16, 512, 1);
  conv3(fTmp, wD4b, W(30), fX5, 10, 512, 8, 16, 512, 1);

  // ---- ConvLSTM x2 ----
  auto lstm_layer = [&](const float* seq, const __bf16* wl, const float* bl, float* hsOut) {
    fillz<<<NB(131072), 256, 0, stream>>>(fH, 131072);
    fillz<<<NB(131072), 256, 0, stream>>>(fC, 131072);
    for (int t = 0; t < 5; ++t) {
      const float* xt = seq + (size_t)t * 131072;
      copy_chan<<<NB(131072), 256, 0, stream>>>(xt, fCat, 2, 512, 1024, 0, 128L);
      copy_chan<<<NB(131072), 256, 0, stream>>>(fH, fCat, 2, 512, 1024, 512, 128L);
      conv3(fCat, wl, bl, fGates, 2, 1024, 8, 16, 2048, 0);
      lstm_point<<<NB(131072), 256, 0, stream>>>(fGates, fH, fC,
                                                 hsOut + (size_t)t * 131072, 2, 512, 128L);
    }
  };
  lstm_layer(fX5, wL0, W(60), fHs0);
  lstm_layer(fHs0, wL1, W(62), fHs1);
  const float* test = fHs1 + (size_t)4 * 131072;   // (2,512,8,16)

  // ---- decoder ----
  auto up_stage = [&](const float* skip, const float* uin, int Cs, int Cu, int Hin, int Win,
                      const __bf16* wa, const float* ba, const __bf16* wb, const float* bb2,
                      int Cout, float* uout) {
    int Ho = 2 * Hin, Wo = 2 * Win;
    long HWo = (long)Ho * Wo;
    up2x_k<<<NB(2L * Cu * HWo), 256, 0, stream>>>(uin, fUp, 2, Cu, Hin, Win);
    copy_chan<<<NB(2L * Cs * HWo), 256, 0, stream>>>(skip, fCatD, 2, Cs, Cs + Cu, 0, HWo);
    copy_chan<<<NB(2L * Cu * HWo), 256, 0, stream>>>(fUp, fCatD, 2, Cu, Cs + Cu, Cs, HWo);
    conv3(fCatD, wa, ba, fTmp, 2, Cs + Cu, Ho, Wo, Cout, 1);
    conv3(fTmp, wb, bb2, uout, 2, Cout, Ho, Wo, Cout, 1);
  };
  up_stage(fX4 + 8L * 262144,  test, 512, 512, 8, 16,  wU1a, W(33), wU1b, W(36), 256, fU);
  up_stage(fX3 + 8L * 524288,  fU,   256, 256, 16, 32, wU2a, W(39), wU2b, W(42), 128, fU);
  up_stage(fX2 + 8L * 1048576, fU,   128, 128, 32, 64, wU3a, W(45), wU3b, W(48), 64, fU);
  up_stage(fX1 + 8L * 2097152, fU,   64, 64, 64, 128,  wU4a, W(51), wU4b, W(54), 64, fU);

  // ---- output: out (2,5,128,256) then test (2,512,8,16) ----
  conv1x1_k<<<NB(327680), 256, 0, stream>>>(fU, W(63), W(64), (float*)d_out,
                                            2, 64, 128, 256, 5);
  copyk<<<NB(131072), 256, 0, stream>>>(test, (float*)d_out + 327680, 131072);
}
